// LogLinearGDNBlock_15358803050898
// MI455X (gfx1250) — matrix operand order repaired and checked by reading, not compiled, not run on hardware
//
#include <hip/hip_runtime.h>
#include <hip/hip_bf16.h>

#define B_     2
#define T_     1024
#define D_     1024
#define H_     8
#define DH     128
#define NL     11
#define CHUNK  64
#define NCHUNK (T_/CHUNK)

typedef __attribute__((ext_vector_type(16))) _Float16 v16h;
typedef __attribute__((ext_vector_type(8)))  float    v8f;

union FragU { v16h v; float4 q[2]; };

__device__ __forceinline__ v8f zero8() {
  v8f z = {0.f,0.f,0.f,0.f,0.f,0.f,0.f,0.f};
  return z;
}

// A fragment (16x32 f16): lane holds K = base..base+15 contiguous (row of A)
__device__ __forceinline__ v16h load_fragA(const _Float16* rowp, int kbase) {
  FragU f;
  f.q[0] = *(const float4*)(rowp + kbase);
  f.q[1] = *(const float4*)(rowp + kbase + 8);
  return f.v;
}
// B fragment (32x16 f16): lane holds K = base..base+15 contiguous (column of B)
__device__ __forceinline__ v16h load_fragB(const _Float16* rowp, int kbase) {
  FragU f;
  f.q[0] = *(const float4*)(rowp + kbase);
  f.q[1] = *(const float4*)(rowp + kbase + 8);
  return f.v;
}
__device__ __forceinline__ v8f wmma32(v16h a, v16h b, v8f c) {
  return __builtin_amdgcn_wmma_f32_16x16x32_f16(false, a, false, b, (short)0, c, false, false);
}
__device__ __forceinline__ float sigmf(float x) { return 1.f / (1.f + expf(-x)); }
__device__ __forceinline__ float siluf(float x) { return x / (1.f + expf(-x)); }

// CDNA5 async global->LDS copy of 16 bytes (tracked by ASYNCcnt).
__device__ __forceinline__ void async_copy_b128(void* ldsDst, const void* gSrc) {
  unsigned lds = (unsigned)(size_t)ldsDst;                  // low 32 bits = LDS offset
  unsigned long long ga = (unsigned long long)(size_t)gSrc; // global vaddr
  asm volatile("global_load_async_to_lds_b128 %0, %1, off"
               :: "v"(lds), "v"(ga) : "memory");
}
__device__ __forceinline__ void async_wait0() {
  asm volatile("s_wait_asynccnt 0x0" ::: "memory");
}

// ---------------------------------------------------------------- cvt f32->f16
__global__ void cvt_f16(const float* __restrict__ x, _Float16* __restrict__ y, int n) {
  int i = blockIdx.x * 256 + threadIdx.x;
  if (i < n) y[i] = (_Float16)x[i];
}

// ------------------------------------------------- transpose W[k][n] -> Wt[n][k] f16
__global__ __launch_bounds__(1024) void transpose_w(const float* __restrict__ W,
                                                    _Float16* __restrict__ Wt,
                                                    int rows, int cols) {
  __shared__ float tile[32][33];
  int bx = blockIdx.x * 32, by = blockIdx.y * 32;
  int tx = threadIdx.x, ty = threadIdx.y;
  tile[ty][tx] = W[(size_t)(by + ty) * cols + bx + tx];
  __syncthreads();
  Wt[(size_t)(bx + ty) * rows + by + tx] = (_Float16)tile[tx][ty];
}

// ---------------------------------------------------------------- WMMA GEMM
// C[M,N] = A[M,K] (f16, row-major) @ Bt[N,K]^T (f16, N-major/K-contig), f32 out.
// Stage tiles with CDNA5 async-to-LDS; compute with v_wmma_f32_16x16x32_f16.
__global__ __launch_bounds__(256) void gemm_f16_wmma(const _Float16* __restrict__ A,
                                                     const _Float16* __restrict__ Bt,
                                                     float* __restrict__ C,
                                                     int M, int N, int K) {
  __shared__ _Float16 As[128 * 40];
  __shared__ _Float16 Bs[128 * 40];
  int tid = threadIdx.x;
  int m0 = blockIdx.y * 128, n0 = blockIdx.x * 128;
  int w = tid >> 5, lane = tid & 31;
  int wm = (w & 3) * 32, wn = (w >> 2) * 64;
  int lr = lane & 15, lh = lane >> 4;

  v8f acc[2][4];
#pragma unroll
  for (int mi = 0; mi < 2; mi++)
#pragma unroll
    for (int ni = 0; ni < 4; ni++) acc[mi][ni] = zero8();

  for (int k0 = 0; k0 < K; k0 += 32) {
#pragma unroll
    for (int i = tid; i < 512; i += 256) {
      int row = i >> 2, cc = (i & 3) * 8;
      const _Float16* ga = A + (size_t)(m0 + row) * K + k0 + cc;
      const _Float16* gb = Bt + (size_t)(n0 + row) * K + k0 + cc;
      async_copy_b128(As + row * 40 + cc, ga);
      async_copy_b128(Bs + row * 40 + cc, gb);
      if (k0 + 32 < K) {
        __builtin_prefetch((const char*)ga + 64, 0, 0);
        __builtin_prefetch((const char*)gb + 64, 0, 0);
      }
    }
    async_wait0();
    __syncthreads();
    v16h a[2], b[4];
#pragma unroll
    for (int mi = 0; mi < 2; mi++) a[mi] = load_fragA(As + (wm + mi * 16 + lr) * 40, lh * 16);
#pragma unroll
    for (int ni = 0; ni < 4; ni++) b[ni] = load_fragB(Bs + (wn + ni * 16 + lr) * 40, lh * 16);
#pragma unroll
    for (int mi = 0; mi < 2; mi++)
#pragma unroll
      for (int ni = 0; ni < 4; ni++) acc[mi][ni] = wmma32(a[mi], b[ni], acc[mi][ni]);
    __syncthreads();
  }
#pragma unroll
  for (int mi = 0; mi < 2; mi++)
#pragma unroll
    for (int ni = 0; ni < 4; ni++)
#pragma unroll
      for (int r = 0; r < 8; r++) {
        int row = m0 + wm + mi * 16 + lh * 8 + r;
        int col = n0 + wn + ni * 16 + lr;
        C[(size_t)row * N + col] = acc[mi][ni][r];
      }
}

// ------------------------------------------ small projections: Wb(8), Wa(8), Wl(88)
__global__ __launch_bounds__(128) void small_proj(const float* __restrict__ x,
                                                  const float* __restrict__ Wb,
                                                  const float* __restrict__ Wa,
                                                  const float* __restrict__ Wl,
                                                  float* __restrict__ blin,
                                                  float* __restrict__ alin,
                                                  float* __restrict__ llin) {
  int row = blockIdx.x;
  int c = threadIdx.x;
  if (c >= 104) return;
  const float* W; int col, nc; float* out;
  if (c < 8)       { W = Wb; col = c;      nc = 8;  out = blin + (size_t)row * 8  + col; }
  else if (c < 16) { W = Wa; col = c - 8;  nc = 8;  out = alin + (size_t)row * 8  + col; }
  else             { W = Wl; col = c - 16; nc = 88; out = llin + (size_t)row * 88 + col; }
  const float* xr = x + (size_t)row * D_;
  float acc = 0.f;
  for (int k = 0; k < D_; k++) acc += xr[k] * W[(size_t)k * nc + col];
  *out = acc;
}

__global__ void lam_sig(const float* __restrict__ a, float* __restrict__ b, int n) {
  int i = blockIdx.x * 256 + threadIdx.x;
  if (i < n) b[i] = sigmf(a[i]);
}

// -------------------------------- causal depthwise conv + SiLU + per-head l2norm
__global__ __launch_bounds__(1024) void conv_silu_norm(
    const float* __restrict__ qp, const float* __restrict__ kp, const float* __restrict__ vp,
    const float* __restrict__ cq, const float* __restrict__ ck, const float* __restrict__ cv,
    _Float16* __restrict__ qh, float* __restrict__ kc, _Float16* __restrict__ kh,
    float* __restrict__ vc) {
  int bt = blockIdx.x;              // b*T + t
  int b = bt >> 10, t = bt & 1023;
  int c = threadIdx.x;              // channel 0..1023
  int h = c >> 7, d = c & 127;
  float aq = 0.f, ak = 0.f, av = 0.f;
#pragma unroll
  for (int j = 0; j < 4; j++) {
    int ts = t - 3 + j;
    if (ts >= 0) {
      size_t idx = ((size_t)b * T_ + ts) * D_ + c;
      aq += qp[idx] * cq[c * 4 + j];
      ak += kp[idx] * ck[c * 4 + j];
      av += vp[idx] * cv[c * 4 + j];
    }
  }
  aq = siluf(aq); ak = siluf(ak); av = siluf(av);

  __shared__ float red[1024];
  red[c] = aq * aq; __syncthreads();
  for (int s = 64; s > 0; s >>= 1) { if (d < s) red[c] += red[c + s]; __syncthreads(); }
  float qn = aq * rsqrtf(red[h * 128] + 1e-6f) * 0.08838834764831845f; // HEAD_DIM^-0.5
  __syncthreads();
  red[c] = ak * ak; __syncthreads();
  for (int s = 64; s > 0; s >>= 1) { if (d < s) red[c] += red[c + s]; __syncthreads(); }
  float kn = ak * rsqrtf(red[h * 128] + 1e-6f);

  size_t bhtd = (((size_t)b * H_ + h) * T_ + t) * DH + d;
  qh[bhtd] = (_Float16)qn;
  kh[bhtd] = (_Float16)kn;
  size_t btc = ((size_t)b * T_ + t) * D_ + c;
  kc[btc] = kn;
  vc[btc] = av;
}

// ------------------------------------- beta / gated-decay cumsum per (b,h)
__global__ __launch_bounds__(1024) void gates_scan(const float* __restrict__ blin,
                                                   const float* __restrict__ alin,
                                                   const float* __restrict__ A_log,
                                                   const float* __restrict__ dt_bias,
                                                   float* __restrict__ beta,
                                                   float* __restrict__ gcs) {
  int bh = blockIdx.x; int b = bh >> 3, h = bh & 7;
  int t = threadIdx.x;
  size_t row = ((size_t)b * T_ + t) * H_ + h;
  beta[(size_t)bh * T_ + t] = sigmf(blin[row]);
  float a = alin[row] + dt_bias[h];
  float sp = (a > 20.f) ? a : log1pf(expf(a));
  float g = -expf(A_log[h]) * sp;
  __shared__ float sm[1024];
  sm[t] = g; __syncthreads();
  for (int off = 1; off < 1024; off <<= 1) {
    float v = sm[t];
    float add = (t >= off) ? sm[t - off] : 0.f;
    __syncthreads();
    sm[t] = v + add;
    __syncthreads();
  }
  gcs[(size_t)bh * T_ + t] = sm[t];
}

// -------------------------- chunked gated delta-rule solve (u), per (b,h)
__global__ __launch_bounds__(256) void chunk_delta(const float* __restrict__ kc,
                                                   const float* __restrict__ vc,
                                                   const float* __restrict__ beta,
                                                   const float* __restrict__ gcs,
                                                   _Float16* __restrict__ uh) {
  extern __shared__ float sm[];
  float* S   = sm;             // 128*128
  float* Ks  = S  + 16384;     // 64*128
  float* Kt  = Ks + 8192;      // 128*64 (transposed)
  float* Us  = Kt + 8192;      // 64*128 rhs -> u
  float* Am  = Us + 8192;      // 64*64
  float* gch = Am + 4096;      // 64
  float* bet = gch + 64;       // 64
  float* qv  = bet + 64;       // 64
  int bh = blockIdx.x; int b = bh >> 3, h = bh & 7;
  int tid = threadIdx.x;
  for (int j = tid; j < 16384; j += 256) S[j] = 0.f;
  float g_prev = 0.f;
  for (int c = 0; c < NCHUNK; ++c) {
    int t0 = c * CHUNK;
    if (tid < 64) {
      gch[tid] = gcs[(size_t)bh * T_ + t0 + tid];
      bet[tid] = beta[(size_t)bh * T_ + t0 + tid];
    }
    for (int j = tid; j < 8192; j += 256) {
      int r = j >> 7, d = j & 127;
      float kv = kc[((size_t)b * T_ + t0 + r) * D_ + h * DH + d];
      Ks[j] = kv;
      Kt[d * 64 + r] = kv;
    }
    __syncthreads();
    // rhs_r = beta_r * (v_r - exp(g_r - g_prevEnd) * k_r @ S)
    for (int j = tid; j < 8192; j += 256) {
      int r = j >> 7, e = j & 127;
      float p = expf(gch[r] - g_prev);
      float dot = 0.f;
      const float* kr = Ks + r * 128;
      for (int d0 = 0; d0 < 128; ++d0) dot += kr[d0] * S[d0 * 128 + e];
      float v = vc[((size_t)b * T_ + t0 + r) * D_ + h * DH + e];
      Us[j] = bet[r] * (v - p * dot);
    }
    // A[r][s] = beta_r * (k_r.k_s) * exp(g_r - g_s)  (strictly lower)
    for (int j = tid; j < 4096; j += 256) {
      int r = j >> 6, s = j & 63;
      float a = 0.f;
      if (s < r) {
        float dot = 0.f;
        const float* kr = Ks + r * 128;
        for (int d0 = 0; d0 < 128; ++d0) dot += kr[d0] * Kt[d0 * 64 + s];
        a = bet[r] * dot * expf(gch[r] - gch[s]);
      }
      Am[j] = a;
    }
    __syncthreads();
    // forward substitution: (I + A) u = rhs
    for (int r = 1; r < 64; ++r) {
      if (tid < 128) {
        float acc = 0.f;
        for (int j = 0; j < r; ++j) acc += Am[r * 64 + j] * Us[j * 128 + tid];
        Us[r * 128 + tid] -= acc;
      }
      __syncthreads();
    }
    for (int j = tid; j < 8192; j += 256) {
      int r = j >> 7, e = j & 127;
      uh[((size_t)bh * T_ + t0 + r) * DH + e] = (_Float16)Us[j];
    }
    float g_last = gch[63];
    if (tid < 64) qv[tid] = expf(g_last - gch[tid]);
    __syncthreads();
    float r0 = expf(g_last - g_prev);
    // S = r0*S + sum_t qv_t * k_t (x) u_t
    for (int j = tid; j < 16384; j += 256) {
      int d = j >> 7, e = j & 127;
      float acc = r0 * S[j];
      const float* kd = Kt + d * 64;
      for (int tt = 0; tt < 64; ++tt) acc += qv[tt] * kd[tt] * Us[tt * 128 + e];
      S[j] = acc;
    }
    g_prev = g_last;
    __syncthreads();
  }
}

// ------------------ log-linear attention readout: o = (q k^T * D * lam[level]) @ u
#define KSP 136
#define UTP 72
#define SCP 72
__global__ __launch_bounds__(256) void loglin_attn(const _Float16* __restrict__ qh,
                                                   const _Float16* __restrict__ kh,
                                                   const _Float16* __restrict__ uh,
                                                   const float* __restrict__ gcs,
                                                   const float* __restrict__ lam,
                                                   float* __restrict__ o) {
  __shared__ _Float16 Ksm[64 * KSP];
  __shared__ _Float16 Ut [128 * UTP];
  __shared__ _Float16 Sc [64 * SCP];
  __shared__ float lamS[64 * 11];
  __shared__ float gct[64];
  __shared__ float gcss[64];
  int bh = blockIdx.x; int b = bh >> 3, h = bh & 7;
  int ct = blockIdx.y; int t0 = ct * CHUNK;
  int tid = threadIdx.x;
  int w = tid >> 5, lane = tid & 31, lr = lane & 15, lh = lane >> 4;
  int rowbase = (w & 3) * 16;
  int colgrp  = (w >> 2);

  for (int j = tid; j < 64 * 11; j += 256) {
    int r = j / 11, l = j % 11;
    lamS[j] = lam[(((size_t)b * T_ + t0 + r) * H_ + h) * NL + l];
  }
  if (tid < 64) gct[tid] = gcs[(size_t)bh * T_ + t0 + tid];

  v8f oacc[4];
#pragma unroll
  for (int ni = 0; ni < 4; ni++) oacc[ni] = zero8();
  const _Float16* qbase = qh + ((size_t)bh * T_) * DH;

  for (int sc = 0; sc <= ct; ++sc) {
    int s0 = sc * CHUNK;
    // K tile: async global->LDS, 16B chunks (64 rows x 16 chunks)
    for (int j = tid; j < 1024; j += 256) {
      int s = j >> 4, c8 = (j & 15) * 8;
      async_copy_b128(Ksm + s * KSP + c8,
                      kh + ((size_t)bh * T_ + s0 + s) * DH + c8);
    }
    // U tile transposed into LDS (for B-operand K-contiguity)
    for (int j = tid; j < 64 * 128; j += 256) {
      int s = j >> 7, d = j & 127;
      Ut[d * UTP + s] = uh[((size_t)bh * T_ + s0 + s) * DH + d];
    }
    if (tid < 64) gcss[tid] = gcs[(size_t)bh * T_ + s0 + tid];
    async_wait0();
    __syncthreads();
    // scores tile (this wave: 16 rows x 32 cols)
    v8f sacc[2]; sacc[0] = zero8(); sacc[1] = zero8();
    const _Float16* qrow = qbase + (size_t)(t0 + rowbase + lr) * DH;
#pragma unroll
    for (int kk = 0; kk < 4; ++kk) {
      v16h aq = load_fragA(qrow, kk * 32 + lh * 16);
#pragma unroll
      for (int ti = 0; ti < 2; ++ti) {
        int scol = colgrp * 32 + ti * 16 + lr;
        v16h bk = load_fragB(Ksm + scol * KSP, kk * 32 + lh * 16);
        sacc[ti] = wmma32(aq, bk, sacc[ti]);
      }
    }
    // decay * Fenwick-level lambda * causal mask, to f16
#pragma unroll
    for (int ti = 0; ti < 2; ++ti) {
#pragma unroll
      for (int r = 0; r < 8; ++r) {
        int m = rowbase + lh * 8 + r;
        int n = colgrp * 32 + ti * 16 + lr;
        int t = t0 + m, s = s0 + n;
        float val = 0.f;
        if (s <= t) {
          int lev = 31 - __builtin_clz((unsigned)(((t + 1) ^ s) | 1));
          if (lev > 10) lev = 10;
          val = sacc[ti][r] * expf(gct[m] - gcss[n]) * lamS[m * 11 + lev];
        }
        Sc[m * SCP + n] = (_Float16)val;
      }
    }
    __syncthreads();
    // o += Sc @ U
#pragma unroll
    for (int kk = 0; kk < 2; ++kk) {
      v16h as = load_fragA(Sc + (rowbase + lr) * SCP, kk * 32 + lh * 16);
#pragma unroll
      for (int ni = 0; ni < 4; ++ni) {
        int e = colgrp * 64 + ni * 16 + lr;
        v16h bu = load_fragB(Ut + e * UTP, kk * 32 + lh * 16);
        oacc[ni] = wmma32(as, bu, oacc[ni]);
      }
    }
    __syncthreads();
  }
#pragma unroll
  for (int ni = 0; ni < 4; ++ni)
#pragma unroll
    for (int r = 0; r < 8; ++r) {
      int t = t0 + rowbase + lh * 8 + r;
      int col = colgrp * 64 + ni * 16 + lr;
      o[((size_t)b * T_ + t) * D_ + h * DH + col] = oacc[ni][r];
    }
}

// ------------------------------------------- gated RMSNorm + SiLU gate, to f16
__global__ __launch_bounds__(1024) void gate_norm(const float* __restrict__ o,
                                                  const float* __restrict__ gp,
                                                  const float* __restrict__ norm_w,
                                                  _Float16* __restrict__ oh) {
  int bt = blockIdx.x;
  int c = threadIdx.x; int h = c >> 7, d = c & 127;
  size_t idx = (size_t)bt * D_ + c;
  float val = o[idx];
  __shared__ float red[1024];
  red[c] = val * val; __syncthreads();
  for (int s = 64; s > 0; s >>= 1) { if (d < s) red[c] += red[c + s]; __syncthreads(); }
  float rms = rsqrtf(red[h * 128] / 128.f + 1e-5f);
  float gt = gp[idx];
  oh[idx] = (_Float16)(val * rms * norm_w[d] * siluf(gt));
}

// ================================================================ launch
extern "C" void kernel_launch(void* const* d_in, const int* in_sizes, int n_in,
                              void* d_out, int out_size, void* d_ws, size_t ws_size,
                              hipStream_t stream) {
  (void)in_sizes; (void)n_in; (void)out_size; (void)ws_size;
  const float* x    = (const float*)d_in[0];
  const float* Wq   = (const float*)d_in[3];
  const float* Wk   = (const float*)d_in[4];
  const float* Wv   = (const float*)d_in[5];
  const float* Wb   = (const float*)d_in[6];
  const float* Wa   = (const float*)d_in[7];
  const float* Wl   = (const float*)d_in[8];
  const float* Wg   = (const float*)d_in[9];
  const float* Wo   = (const float*)d_in[10];
  const float* cq   = (const float*)d_in[11];
  const float* ck   = (const float*)d_in[12];
  const float* cv   = (const float*)d_in[13];
  const float* Alog = (const float*)d_in[14];
  const float* dtb  = (const float*)d_in[15];
  const float* nw   = (const float*)d_in[16];

  char* ws = (char*)d_ws;
  size_t off = 0;
  auto alloc = [&](size_t bytes) -> char* {
    char* p = ws + off;
    off += (bytes + 255) & ~(size_t)255;
    return p;
  };
  const size_t NE = (size_t)B_ * T_ * D_;   // 2M elements
  _Float16* xh   = (_Float16*)alloc(NE * 2);
  _Float16* wtq  = (_Float16*)alloc((size_t)D_ * D_ * 2);
  _Float16* wtk  = (_Float16*)alloc((size_t)D_ * D_ * 2);
  _Float16* wtv  = (_Float16*)alloc((size_t)D_ * D_ * 2);
  _Float16* wtg  = (_Float16*)alloc((size_t)D_ * D_ * 2);
  _Float16* wto  = (_Float16*)alloc((size_t)D_ * D_ * 2);
  float* qp   = (float*)alloc(NE * 4);
  float* kp   = (float*)alloc(NE * 4);
  float* vp   = (float*)alloc(NE * 4);
  float* gp   = (float*)alloc(NE * 4);
  float* blin = (float*)alloc((size_t)B_ * T_ * 8 * 4);
  float* alin = (float*)alloc((size_t)B_ * T_ * 8 * 4);
  float* llin = (float*)alloc((size_t)B_ * T_ * 88 * 4);
  float* lamv = (float*)alloc((size_t)B_ * T_ * 88 * 4);
  float* betaP= (float*)alloc((size_t)B_ * H_ * T_ * 4);
  float* gcsP = (float*)alloc((size_t)B_ * H_ * T_ * 4);
  float* kc   = (float*)alloc(NE * 4);
  float* vc   = (float*)alloc(NE * 4);
  _Float16* qhp = (_Float16*)alloc(NE * 2);
  _Float16* khp = (_Float16*)alloc(NE * 2);
  _Float16* uhp = (_Float16*)alloc(NE * 2);
  float* oP   = (float*)alloc(NE * 4);
  _Float16* ohp = (_Float16*)alloc(NE * 2);

  cvt_f16<<<(int)(NE / 256), 256, 0, stream>>>(x, xh, (int)NE);

  dim3 tb(32, 32), tg(32, 32);
  transpose_w<<<tg, tb, 0, stream>>>(Wq, wtq, D_, D_);
  transpose_w<<<tg, tb, 0, stream>>>(Wk, wtk, D_, D_);
  transpose_w<<<tg, tb, 0, stream>>>(Wv, wtv, D_, D_);
  transpose_w<<<tg, tb, 0, stream>>>(Wg, wtg, D_, D_);
  transpose_w<<<tg, tb, 0, stream>>>(Wo, wto, D_, D_);

  dim3 gg(D_ / 128, (B_ * T_) / 128);   // (8, 16)
  gemm_f16_wmma<<<gg, 256, 0, stream>>>(xh, wtq, qp, B_ * T_, D_, D_);
  gemm_f16_wmma<<<gg, 256, 0, stream>>>(xh, wtk, kp, B_ * T_, D_, D_);
  gemm_f16_wmma<<<gg, 256, 0, stream>>>(xh, wtv, vp, B_ * T_, D_, D_);
  gemm_f16_wmma<<<gg, 256, 0, stream>>>(xh, wtg, gp, B_ * T_, D_, D_);

  small_proj<<<B_ * T_, 128, 0, stream>>>(x, Wb, Wa, Wl, blin, alin, llin);
  lam_sig<<<(B_ * T_ * 88 + 255) / 256, 256, 0, stream>>>(llin, lamv, B_ * T_ * 88);

  conv_silu_norm<<<B_ * T_, 1024, 0, stream>>>(qp, kp, vp, cq, ck, cv, qhp, kc, khp, vc);
  gates_scan<<<B_ * H_, 1024, 0, stream>>>(blin, alin, Alog, dtb, betaP, gcsP);

  size_t smem_cd = (size_t)(16384 + 8192 + 8192 + 8192 + 4096 + 64 + 64 + 64) * 4;
  hipFuncSetAttribute(reinterpret_cast<const void*>(chunk_delta),
                      hipFuncAttributeMaxDynamicSharedMemorySize, (int)smem_cd);
  chunk_delta<<<B_ * H_, 256, smem_cd, stream>>>(kc, vc, betaP, gcsP, uhp);

  dim3 ag(B_ * H_, NCHUNK);             // (16, 16)
  loglin_attn<<<ag, 256, 0, stream>>>(qhp, khp, uhp, gcsP, lamv, oP);

  gate_norm<<<B_ * T_, 1024, 0, stream>>>(oP, gp, nw, ohp);

  gemm_f16_wmma<<<gg, 256, 0, stream>>>(ohp, wto, (float*)d_out, B_ * T_, D_, D_);
}
